// Model_59201829208523
// MI455X (gfx1250) — matrix operand order repaired and checked by reading, not compiled, not run on hardware
//
#include <hip/hip_runtime.h>
#include <hip/hip_bf16.h>
#include <math.h>

// ---------------------------------------------------------------------------
// Types for CDNA5 WMMA
// ---------------------------------------------------------------------------
typedef __attribute__((ext_vector_type(16))) _Float16 v16h;
typedef __attribute__((ext_vector_type(8)))  float    v8f;

// ---------------------------------------------------------------------------
// Model dimensions (fixed by the reference)
// ---------------------------------------------------------------------------
#define B_SZ   16
#define CTX_T  200
#define D_EMB  512
#define H_N    8
#define HD     64           // D_EMB / H_N
#define NX_L   8
#define VOCAB  50259
#define FF4    2048         // 4*D
#define M_ROWS (B_SZ * CTX_T)   // 3200

// GEMM tiling: 128x64 block tile, 8 waves (wave32), K staged 64 at a time
#define TM     128
#define TN     64
#define KT     64
#define LDK    72           // LDS row stride in halves (144B, keeps 16B slices)

// ---------------------------------------------------------------------------
// Fragment loads from LDS (per CDNA5 ISA §7.12.2 16-bit layouts)
// ---------------------------------------------------------------------------
__device__ __forceinline__ v16h frag_a(const _Float16* row, int lane) {
    // A 16x32: lane%16 = M ; halves: K = g*8..g*8+7 then +16  (g = lane>>4)
    int g = (lane >> 4) * 8;
    union { v16h h; float4 f[2]; } u;
    u.f[0] = *reinterpret_cast<const float4*>(row + g);
    u.f[1] = *reinterpret_cast<const float4*>(row + 16 + g);
    return u.h;
}

__device__ __forceinline__ v16h frag_b(const _Float16* row, int lane) {
    // B 32x16 staged transposed as [N][K]: lane%16 = N ; halves: K = g*8..g*8+7 then +16  (g = lane>>4)
    int g = (lane >> 4) * 8;
    union { v16h h; float4 f[2]; } u;
    u.f[0] = *reinterpret_cast<const float4*>(row + g);
    u.f[1] = *reinterpret_cast<const float4*>(row + 16 + g);
    return u.h;
}

// ---------------------------------------------------------------------------
// Tiled WMMA GEMM:  C[M,N] = A[M,K] @ W[K,N] (+bias) (relu?) (+resid)
// fp32 in/out, f16 multiply, f32 accumulate via v_wmma_f32_16x16x32_f16
// K must be a multiple of 64 (true for all call sites: 512 / 2048).
// Out-of-range rows/cols are CLAMPED at load (garbage only pollutes
// outputs that the bounds-checked epilogue never stores).
// ---------------------------------------------------------------------------
__global__ __launch_bounds__(256) void gemm_wmma(
    const float* __restrict__ A, const float* __restrict__ W,
    const float* __restrict__ bias, const float* __restrict__ resid,
    float* __restrict__ C, int M, int N, int K, int relu)
{
    __shared__ alignas(16) _Float16 As[TM * LDK];
    __shared__ alignas(16) _Float16 Bs[TN * LDK];

    const int tid  = threadIdx.x;
    const int lane = tid & 31;
    const int wave = tid >> 5;
    const int wm   = (wave >> 1) * 32;      // 0,32,64,96
    const int wn   = (wave & 1)  * 32;      // 0,32
    const int m0   = blockIdx.y * TM;
    const int n0   = blockIdx.x * TN;

    v8f acc[2][2] = {};

    // A staging: thread -> (row ar, 32 contiguous k) ; 128 rows x 64 k
    const int ar = tid >> 1;                 // 0..127
    const int ac = (tid & 1) * 32;           // 0 or 32
    const int arow = (m0 + ar < M) ? (m0 + ar) : (M - 1);   // clamp
    // B staging: thread -> (col nr, 16 k values read strided) ; 64 cols x 64 k
    const int nr = tid >> 2;                 // 0..63
    const int kc = (tid & 3) * 16;           // 0,16,32,48
    const int ncol = (n0 + nr < N) ? (n0 + nr) : (N - 1);   // clamp

    for (int k0 = 0; k0 < K; k0 += KT) {
        __syncthreads();
        // ---- stage A tile: 8 float4 loads -> cvt -> 4 ds_store_b128 ----
        {
            const float* src = A + (size_t)arow * K + k0 + ac;
            float fa[32];
#pragma unroll
            for (int j = 0; j < 8; ++j) {
                float4 t = *reinterpret_cast<const float4*>(src + j * 4);
                fa[j * 4 + 0] = t.x; fa[j * 4 + 1] = t.y;
                fa[j * 4 + 2] = t.z; fa[j * 4 + 3] = t.w;
            }
            if (k0 + KT < K) __builtin_prefetch(src + KT, 0, 0);
            union { _Float16 h[32]; float4 f4[4]; } ua;
#pragma unroll
            for (int j = 0; j < 32; ++j) ua.h[j] = (_Float16)fa[j];
            float4* dst4 = reinterpret_cast<float4*>(As + ar * LDK + ac);
#pragma unroll
            for (int j = 0; j < 4; ++j) dst4[j] = ua.f4[j];
        }
        // ---- stage B tile transposed: Bs[n][k] = W[k0+k][n0+n] ----
        {
            float fb[16];
#pragma unroll
            for (int j = 0; j < 16; ++j)
                fb[j] = W[(size_t)(k0 + kc + j) * N + ncol];
            union { _Float16 h[16]; float4 f4[2]; } ub;
#pragma unroll
            for (int j = 0; j < 16; ++j) ub.h[j] = (_Float16)fb[j];
            float4* dst4 = reinterpret_cast<float4*>(Bs + nr * LDK + kc);
            dst4[0] = ub.f4[0];
            dst4[1] = ub.f4[1];
        }
        __syncthreads();

        // ---- two WMMA K-steps per stage ----
#pragma unroll
        for (int ks = 0; ks < KT; ks += 32) {
            v16h af[2], bf[2];
#pragma unroll
            for (int i = 0; i < 2; ++i) {
                af[i] = frag_a(As + (wm + i * 16 + (lane & 15)) * LDK + ks, lane);
                bf[i] = frag_b(Bs + (wn + i * 16 + (lane & 15)) * LDK + ks, lane);
            }
#pragma unroll
            for (int i = 0; i < 2; ++i)
#pragma unroll
                for (int j = 0; j < 2; ++j)
                    acc[i][j] = __builtin_amdgcn_wmma_f32_16x16x32_f16(
                        false, af[i], false, bf[j], (short)0, acc[i][j], false, false);
        }
    }

    // ---- epilogue: bias / relu / residual, bounds-checked store ----
#pragma unroll
    for (int i = 0; i < 2; ++i) {
#pragma unroll
        for (int j = 0; j < 2; ++j) {
            const int col = n0 + wn + j * 16 + (lane & 15);
            if (col >= N) continue;
#pragma unroll
            for (int r = 0; r < 8; ++r) {
                const int row = m0 + wm + i * 16 + (lane >> 4) * 8 + r;
                if (row >= M) continue;
                float v = acc[i][j][r];
                if (bias)  v += bias[col];
                if (relu)  v = fmaxf(v, 0.f);
                if (resid) v += resid[(size_t)row * N + col];
                C[(size_t)row * N + col] = v;
            }
        }
    }
}

// ---------------------------------------------------------------------------
// Embedding + sinusoidal PE (PE computed inline; it is not a kernel input)
// ---------------------------------------------------------------------------
__global__ void embed_kernel(const int* __restrict__ tok,
                             const float* __restrict__ emb,
                             float* __restrict__ Y, int T, int total)
{
    int i = blockIdx.x * blockDim.x + threadIdx.x;
    if (i >= total) return;
    int d  = i % D_EMB;
    int bt = i / D_EMB;
    int t  = bt % T;
    int i2 = d & ~1;
    float div = __expf(-(float)i2 * (9.210340371976184f / (float)D_EMB)); // ln(1e4)/D
    float a = (float)t * div;
    float pe = (d & 1) ? __cosf(a) : __sinf(a);
    Y[i] = emb[(size_t)tok[bt] * D_EMB + d] + pe;
}

// ---------------------------------------------------------------------------
// Zero whole rows where mask[row]==0 (faithful to masked_fill on q/k)
// ---------------------------------------------------------------------------
__global__ void maskzero_kernel(float* __restrict__ buf,
                                const int* __restrict__ mask, int total)
{
    int i = blockIdx.x * blockDim.x + threadIdx.x;
    if (i >= total) return;
    if (mask[i / D_EMB] == 0) buf[i] = 0.f;
}

// ---------------------------------------------------------------------------
// LayerNorm: one wave per row of D=512
// ---------------------------------------------------------------------------
__global__ __launch_bounds__(256) void layernorm_kernel(
    const float* __restrict__ X, const float* __restrict__ gs,
    const float* __restrict__ gb, float* __restrict__ Y, int M)
{
    const int lane = threadIdx.x & 31;
    const int w    = threadIdx.x >> 5;
    const int row  = blockIdx.x * 8 + w;
    if (row >= M) return;
    const float* x = X + (size_t)row * D_EMB;

    float v[16];
    float s = 0.f;
#pragma unroll
    for (int i = 0; i < 16; ++i) { v[i] = x[lane + i * 32]; s += v[i]; }
#pragma unroll
    for (int o = 16; o; o >>= 1) s += __shfl_xor(s, o, 32);
    const float mean = s * (1.f / D_EMB);

    float q = 0.f;
#pragma unroll
    for (int i = 0; i < 16; ++i) { float d = v[i] - mean; q += d * d; }
#pragma unroll
    for (int o = 16; o; o >>= 1) q += __shfl_xor(q, o, 32);
    const float rstd = rsqrtf(q * (1.f / D_EMB) + 1e-5f);

    float* y = Y + (size_t)row * D_EMB;
#pragma unroll
    for (int i = 0; i < 16; ++i) {
        int c = lane + i * 32;
        y[c] = (v[i] - mean) * rstd * gs[c] + gb[c];
    }
}

// ---------------------------------------------------------------------------
// Attention: one wave per (b, h, q) row. scores -> softmax -> P @ V
// Q/K/V are [B*T, D] with head h at columns h*64..h*64+63.
// ---------------------------------------------------------------------------
__global__ __launch_bounds__(256) void attn_kernel(
    const float* __restrict__ Q, const float* __restrict__ Kb,
    const float* __restrict__ Vb, float* __restrict__ O,
    float scale, int causal)
{
    __shared__ alignas(16) float qs[8][HD];
    __shared__ float ps[8][CTX_T + 8];

    const int lane = threadIdx.x & 31;
    const int w    = threadIdx.x >> 5;
    const int row  = blockIdx.x * 8 + w;          // grid sized exactly: B*H*T/8
    const int b    = row / (H_N * CTX_T);
    const int rem  = row % (H_N * CTX_T);
    const int hh   = rem / CTX_T;
    const int qp   = rem % CTX_T;

    const size_t base = (size_t)(b * CTX_T) * D_EMB + hh * HD;

    // stage q row
    const float* qrow = Q + base + (size_t)qp * D_EMB;
    qs[w][lane]      = qrow[lane];
    qs[w][lane + 32] = qrow[lane + 32];
    __syncthreads();

    // scores: 32 keys at a time, lane-parallel, float4 dot products
    for (int k0 = 0; k0 < CTX_T; k0 += 32) {
        int kk = k0 + lane;
        if (kk < CTX_T) {
            float s = -INFINITY;
            if (!causal || kk <= qp) {
                const float* krow = Kb + base + (size_t)kk * D_EMB;
                float acc = 0.f;
#pragma unroll
                for (int d = 0; d < HD; d += 4) {
                    float4 qv = *reinterpret_cast<const float4*>(&qs[w][d]);
                    float4 kv = *reinterpret_cast<const float4*>(krow + d);
                    acc += qv.x * kv.x + qv.y * kv.y + qv.z * kv.z + qv.w * kv.w;
                }
                s = acc * scale;
            }
            ps[w][kk] = s;
        }
    }

    // softmax over ps[w][0..T)
    float mx = -INFINITY;
    for (int kk = lane; kk < CTX_T; kk += 32) mx = fmaxf(mx, ps[w][kk]);
#pragma unroll
    for (int o = 16; o; o >>= 1) mx = fmaxf(mx, __shfl_xor(mx, o, 32));
    float sum = 0.f;
    for (int kk = lane; kk < CTX_T; kk += 32) {
        float e = __expf(ps[w][kk] - mx);
        ps[w][kk] = e;
        sum += e;
    }
#pragma unroll
    for (int o = 16; o; o >>= 1) sum += __shfl_xor(sum, o, 32);
    const float inv = 1.f / sum;
    __syncthreads();

    // O = P @ V (lanes parallel over head dim, coalesced V reads)
    float a0 = 0.f, a1 = 0.f;
    for (int kk = 0; kk < CTX_T; ++kk) {
        float p = ps[w][kk];
        const float* vrow = Vb + base + (size_t)kk * D_EMB;
        a0 += p * vrow[lane];
        a1 += p * vrow[lane + 32];
    }
    float* orow = O + base + (size_t)qp * D_EMB;
    orow[lane]      = a0 * inv;
    orow[lane + 32] = a1 * inv;
}

// ---------------------------------------------------------------------------
// Host orchestration
// ---------------------------------------------------------------------------
extern "C" void kernel_launch(void* const* d_in, const int* in_sizes, int n_in,
                              void* d_out, int out_size, void* d_ws, size_t ws_size,
                              hipStream_t stream)
{
    (void)in_sizes; (void)n_in; (void)out_size; (void)ws_size;

    // ---- inputs (setup_inputs order) ----
    const int*   x        = (const int*)  d_in[0];
    const int*   src_mask = (const int*)  d_in[1];
    const int*   predict  = (const int*)  d_in[2];
    const int*   pred_mask= (const int*)  d_in[3];
    const float* tok_emb  = (const float*)d_in[4];
    const float* e_wq  = (const float*)d_in[5];
    const float* e_wk  = (const float*)d_in[6];
    const float* e_wv  = (const float*)d_in[7];
    const float* e_pw  = (const float*)d_in[8];
    const float* e_pb  = (const float*)d_in[9];
    const float* e_w1  = (const float*)d_in[10];
    const float* e_b1  = (const float*)d_in[11];
    const float* e_w2  = (const float*)d_in[12];
    const float* e_b2  = (const float*)d_in[13];
    const float* e_ln1s = (const float*)d_in[14];
    const float* e_ln1b = (const float*)d_in[15];
    const float* e_ln2s = (const float*)d_in[16];
    const float* e_ln2b = (const float*)d_in[17];
    const float* d_sq  = (const float*)d_in[18];
    const float* d_sk  = (const float*)d_in[19];
    const float* d_sv  = (const float*)d_in[20];
    const float* d_spw = (const float*)d_in[21];
    const float* d_spb = (const float*)d_in[22];
    const float* d_cq  = (const float*)d_in[23];
    const float* d_ck  = (const float*)d_in[24];
    const float* d_cv  = (const float*)d_in[25];
    const float* d_cpw = (const float*)d_in[26];
    const float* d_cpb = (const float*)d_in[27];
    const float* d_w1  = (const float*)d_in[28];
    const float* d_b1  = (const float*)d_in[29];
    const float* d_w2  = (const float*)d_in[30];
    const float* d_b2  = (const float*)d_in[31];
    const float* d_ln1s = (const float*)d_in[32];
    const float* d_ln1b = (const float*)d_in[33];
    const float* d_ln2s = (const float*)d_in[34];
    const float* d_ln2b = (const float*)d_in[35];
    const float* d_ln3s = (const float*)d_in[36];
    const float* d_ln3b = (const float*)d_in[37];
    const float* lnf_s  = (const float*)d_in[38];
    const float* lnf_b  = (const float*)d_in[39];
    const float* head_w = (const float*)d_in[40];
    const float* head_b = (const float*)d_in[41];
    float* out = (float*)d_out;

    // ---- workspace carve-up (floats) ----
    const size_t MD = (size_t)M_ROWS * D_EMB;         // 1,638,400
    float* wsf = (float*)d_ws;
    float* h   = wsf;            // encoder stream / mem
    float* dec = h   + MD;       // decoder stream
    float* lnb = dec + MD;       // layernorm output
    float* qb  = lnb + MD;
    float* kb  = qb  + MD;
    float* vb  = kb  + MD;
    float* ab  = vb  + MD;       // attention output
    float* ff  = ab  + MD;       // M x 4D FFN hidden

    const float SCALE = 0.044194173824159216f;        // 512^-0.5

    const int DD  = D_EMB * D_EMB;
    const int DF  = D_EMB * FF4;

    auto gemm = [&](const float* Ap, const float* Wp, const float* bp,
                    const float* rp, float* Cp, int Mq, int Nq, int Kq, int relu) {
        dim3 g((Nq + TN - 1) / TN, (Mq + TM - 1) / TM);
        gemm_wmma<<<g, dim3(256), 0, stream>>>(Ap, Wp, bp, rp, Cp, Mq, Nq, Kq, relu);
    };
    auto ln = [&](const float* Xp, const float* sp, const float* bp, float* Yp) {
        layernorm_kernel<<<dim3(M_ROWS / 8), dim3(256), 0, stream>>>(Xp, sp, bp, Yp, M_ROWS);
    };
    auto mz = [&](float* buf, const int* mask) {
        int total = (int)MD;
        maskzero_kernel<<<dim3((total + 255) / 256), dim3(256), 0, stream>>>(buf, mask, total);
    };
    auto attn = [&](int causal) {
        attn_kernel<<<dim3((B_SZ * H_N * CTX_T) / 8), dim3(256), 0, stream>>>(
            qb, kb, vb, ab, SCALE, causal);
    };

    // ================= Encoder =================
    {
        int total = (int)MD;
        embed_kernel<<<dim3((total + 255) / 256), dim3(256), 0, stream>>>(
            x, tok_emb, h, CTX_T, total);
    }
    for (int i = 0; i < NX_L; ++i) {
        ln(h, e_ln1s + i * D_EMB, e_ln1b + i * D_EMB, lnb);
        gemm(lnb, e_wq + (size_t)i * DD, nullptr, nullptr, qb, M_ROWS, D_EMB, D_EMB, 0);
        gemm(lnb, e_wk + (size_t)i * DD, nullptr, nullptr, kb, M_ROWS, D_EMB, D_EMB, 0);
        gemm(lnb, e_wv + (size_t)i * DD, nullptr, nullptr, vb, M_ROWS, D_EMB, D_EMB, 0);
        mz(qb, src_mask);
        mz(kb, src_mask);
        attn(/*causal=*/0);
        gemm(ab, e_pw + (size_t)i * DD, e_pb + i * D_EMB, h, h, M_ROWS, D_EMB, D_EMB, 0);
        ln(h, e_ln2s + i * D_EMB, e_ln2b + i * D_EMB, lnb);
        gemm(lnb, e_w1 + (size_t)i * DF, e_b1 + i * FF4, nullptr, ff, M_ROWS, FF4, D_EMB, 1);
        gemm(ff, e_w2 + (size_t)i * DF, e_b2 + i * D_EMB, h, h, M_ROWS, D_EMB, FF4, 0);
    }

    // ================= Decoder =================
    {
        int total = (int)MD;
        embed_kernel<<<dim3((total + 255) / 256), dim3(256), 0, stream>>>(
            predict, tok_emb, dec, CTX_T, total);
    }
    for (int i = 0; i < NX_L; ++i) {
        // masked causal self-attention
        ln(dec, d_ln1s + i * D_EMB, d_ln1b + i * D_EMB, lnb);
        gemm(lnb, d_sq + (size_t)i * DD, nullptr, nullptr, qb, M_ROWS, D_EMB, D_EMB, 0);
        gemm(lnb, d_sk + (size_t)i * DD, nullptr, nullptr, kb, M_ROWS, D_EMB, D_EMB, 0);
        gemm(lnb, d_sv + (size_t)i * DD, nullptr, nullptr, vb, M_ROWS, D_EMB, D_EMB, 0);
        mz(qb, pred_mask);
        mz(kb, pred_mask);
        attn(/*causal=*/1);
        gemm(ab, d_spw + (size_t)i * DD, d_spb + i * D_EMB, dec, dec, M_ROWS, D_EMB, D_EMB, 0);

        // cross-attention (faithful to reference: K from mem, V from decoder x; causal)
        ln(dec, d_ln2s + i * D_EMB, d_ln2b + i * D_EMB, lnb);
        gemm(lnb, d_cq + (size_t)i * DD, nullptr, nullptr, qb, M_ROWS, D_EMB, D_EMB, 0);
        gemm(h,   d_ck + (size_t)i * DD, nullptr, nullptr, kb, M_ROWS, D_EMB, D_EMB, 0);
        gemm(lnb, d_cv + (size_t)i * DD, nullptr, nullptr, vb, M_ROWS, D_EMB, D_EMB, 0);
        mz(kb, src_mask);
        attn(/*causal=*/1);
        gemm(ab, d_cpw + (size_t)i * DD, d_cpb + i * D_EMB, dec, dec, M_ROWS, D_EMB, D_EMB, 0);

        // FFN
        ln(dec, d_ln3s + i * D_EMB, d_ln3b + i * D_EMB, lnb);
        gemm(lnb, d_w1 + (size_t)i * DF, d_b1 + i * FF4, nullptr, ff, M_ROWS, FF4, D_EMB, 1);
        gemm(ff, d_w2 + (size_t)i * DF, d_b2 + i * D_EMB, dec, dec, M_ROWS, D_EMB, FF4, 0);
    }

    // ================= Head =================
    ln(dec, lnf_s, lnf_b, lnb);
    gemm(lnb, head_w, head_b, nullptr, out, M_ROWS, VOCAB, D_EMB, 0);
}
